// DeepGemmFP8BlockScalesRefFusedMoE_47562467836581
// MI455X (gfx1250) — 8-bit matrix operand order repaired and checked by reading, compiled, not run on hardware
//
#include <hip/hip_runtime.h>
#include <hip/hip_bf16.h>
#include <stdint.h>

#define T_TOK 1024
#define H_DIM 1024
#define I_DIM 768
#define I2_DIM 1536
#define E_EXP 8
#define KB_H 8
#define KB_I 6
#define FP8MAX 448.0f

typedef __attribute__((ext_vector_type(16))) int   v16i_t;
typedef __attribute__((ext_vector_type(8)))  float v8f_t;

__device__ __forceinline__ unsigned f32_to_fp8(float x) {
    unsigned u = __float_as_uint(x);
    unsigned sign = (u >> 31) << 7;
    float ax = fabsf(x);
    if (!(ax > 0.0f)) return sign;
    if (ax >= FP8MAX) return sign | 0x7Eu;
    unsigned au = __float_as_uint(ax);
    int  ex   = (int)((au >> 23) & 0xFF) - 127;
    unsigned mant = au & 0x7FFFFFu;
    unsigned enc;
    if (ex < -6) {
        int shift = 20 + (-6 - ex);
        if (shift > 24) return sign;
        unsigned full = mant | 0x800000u;
        unsigned r    = full >> shift;
        unsigned rem  = full & ((1u << shift) - 1u);
        unsigned half = 1u << (shift - 1);
        if (rem > half || (rem == half && (r & 1u))) r++;
        enc = r;
    } else {
        unsigned r   = mant >> 20;
        unsigned rem = mant & 0xFFFFFu;
        if (rem > 0x80000u || (rem == 0x80000u && (r & 1u))) r++;
        enc = ((unsigned)(ex + 7) << 3) + r;
        if (enc > 0x7Eu) enc = 0x7Eu;
    }
    return sign | enc;
}

__global__ void moe_router_kernel(const float* __restrict__ logits,
                                  float* __restrict__ Wr) {
    int t = blockIdx.x * 256 + threadIdx.x;
    if (t >= T_TOK) return;
    float p[E_EXP];
    float m = -1e30f;
#pragma unroll
    for (int e = 0; e < E_EXP; ++e) { p[e] = logits[t * E_EXP + e]; m = fmaxf(m, p[e]); }
    float s = 0.f;
#pragma unroll
    for (int e = 0; e < E_EXP; ++e) { p[e] = __expf(p[e] - m); s += p[e]; }
    float inv = 1.0f / s;
#pragma unroll
    for (int e = 0; e < E_EXP; ++e) p[e] *= inv;
    int i1 = 0;
#pragma unroll
    for (int e = 1; e < E_EXP; ++e) if (p[e] > p[i1]) i1 = e;
    int i2 = (i1 == 0) ? 1 : 0;
#pragma unroll
    for (int e = 0; e < E_EXP; ++e) if (e != i1 && p[e] > p[i2]) i2 = e;
#pragma unroll
    for (int e = 0; e < E_EXP; ++e)
        Wr[t * E_EXP + e] = (e == i1 || e == i2) ? p[e] : 0.0f;
}

__global__ void cvt_w13_kernel(const float* __restrict__ w, uint8_t* __restrict__ wT) {
    size_t idx = (size_t)blockIdx.x * 256 + threadIdx.x;
    const size_t total = (size_t)E_EXP * I2_DIM * H_DIM;
    if (idx >= total) return;
    int k = (int)(idx % H_DIM);
    size_t r = idx / H_DIM;
    int o = (int)(r % I2_DIM);
    int e = (int)(r / I2_DIM);
    wT[((size_t)e * H_DIM + k) * I2_DIM + o] = (uint8_t)f32_to_fp8(w[idx]);
}

__global__ void cvt_w2_kernel(const float* __restrict__ w, uint8_t* __restrict__ wT) {
    size_t idx = (size_t)blockIdx.x * 256 + threadIdx.x;
    const size_t total = (size_t)E_EXP * H_DIM * I_DIM;
    if (idx >= total) return;
    int i = (int)(idx % I_DIM);
    size_t r = idx / I_DIM;
    int h = (int)(r % H_DIM);
    int e = (int)(r / H_DIM);
    wT[((size_t)e * I_DIM + i) * H_DIM + h] = (uint8_t)f32_to_fp8(w[idx]);
}

__global__ void quant_rows_kernel(const float* __restrict__ x,
                                  uint8_t* __restrict__ q,
                                  float* __restrict__ scales, int nrows) {
    int row  = blockIdx.x * 8 + (threadIdx.x >> 5);
    int lane = threadIdx.x & 31;
    if (row >= nrows) return;
    const float4 v = *(const float4*)(x + (size_t)row * 128 + lane * 4);
    float am = fmaxf(fmaxf(fabsf(v.x), fabsf(v.y)), fmaxf(fabsf(v.z), fabsf(v.w)));
#pragma unroll
    for (int off = 16; off > 0; off >>= 1)
        am = fmaxf(am, __shfl_xor(am, off, 32));
    am = fmaxf(am, 1e-10f);
    int ex; float mant = frexpf(am / FP8MAX, &ex);
    float scale = ldexpf(1.0f, (mant > 0.5f) ? ex : ex - 1);
    float rs = 1.0f / scale;
    unsigned packed = f32_to_fp8(v.x * rs)
                    | (f32_to_fp8(v.y * rs) << 8)
                    | (f32_to_fp8(v.z * rs) << 16)
                    | (f32_to_fp8(v.w * rs) << 24);
    *(unsigned*)(q + (size_t)row * 128 + lane * 4) = packed;
    if (lane == 0) scales[row] = scale;
}

__device__ __forceinline__ v16i_t load_a_frag(const uint8_t* __restrict__ rowPtr,
                                              int kbase, int lane) {
    v16i_t A;
    int kb_off = kbase + ((lane >> 4) << 3);
#pragma unroll
    for (int j = 0; j < 16; ++j) {
        int jj = j & 7;
        int k = kb_off + ((j >> 3) << 6) + ((jj >> 1) << 4) + ((jj & 1) << 2);
        A[j] = *(const int*)(rowPtr + k);
    }
    return A;
}
__device__ __forceinline__ v16i_t load_b_frag(const uint8_t* __restrict__ colPtr,
                                              int kbase, size_t nstride, int lane) {
    v16i_t B;
    int kb_off = kbase + ((lane >> 4) << 3);
    const uint8_t* cp = colPtr + (lane & 15);
#pragma unroll
    for (int j = 0; j < 16; ++j) {
        int jj = j & 7;
        int k = kb_off + ((j >> 3) << 6) + ((jj >> 1) << 4) + ((jj & 1) << 2);
        const uint8_t* p = cp + (size_t)k * nstride;
        B[j] = (int)((unsigned)p[0] | ((unsigned)p[nstride] << 8) |
                     ((unsigned)p[2 * nstride] << 16) | ((unsigned)p[3 * nstride] << 24));
    }
    return B;
}

__global__ void __launch_bounds__(128, 1)
moe_gemm1_kernel(const uint8_t* __restrict__ act,
                 const float* __restrict__ ascale,
                 const uint8_t* __restrict__ w13T,
                 const float* __restrict__ w13s,
                 float* __restrict__ actbuf) {
    const int lane = threadIdx.x & 31;
    const int wave = threadIdx.x >> 5;
    const int it = blockIdx.x * 4 + wave;
    const int t0 = blockIdx.y * 16;
    const int e  = blockIdx.z;
    const int i0 = it * 16;

    const uint8_t* aRow  = act + (size_t)(t0 + (lane & 15)) * H_DIM;
    const uint8_t* bBase = w13T + (size_t)e * H_DIM * I2_DIM;
    const uint8_t* bVal  = bBase + i0;
    const uint8_t* bGate = bBase + i0 + I_DIM;
    const int obv = i0 >> 7;
    const int obg = (i0 + I_DIM) >> 7;
    const int mbase = t0 + ((lane >> 4) << 3);
    const float* saBase = ascale + mbase * KB_H;
    const float* sbvBase = w13s + e * 96 + obv * 8;
    const float* sbgBase = w13s + e * 96 + obg * 8;

    v8f_t accV = {}, accG = {};
    const v8f_t zero = {};
#pragma unroll 1
    for (int kb = 0; kb < KB_H; ++kb) {
        if (kb + 1 < KB_H) {
            __builtin_prefetch(bVal + (size_t)(kb + 1) * 128 * I2_DIM + lane * I2_DIM, 0, 1);
            __builtin_prefetch(bGate + (size_t)(kb + 1) * 128 * I2_DIM + lane * I2_DIM, 0, 1);
        }
        v16i_t A  = load_a_frag(aRow, kb * 128, lane);
        v16i_t Bv = load_b_frag(bVal,  kb * 128, I2_DIM, lane);
        v16i_t Bg = load_b_frag(bGate, kb * 128, I2_DIM, lane);
        float sbv = sbvBase[kb];
        float sbg = sbgBase[kb];
        float sa[8];
#pragma unroll
        for (int r = 0; r < 8; ++r) sa[r] = saBase[r * KB_H + kb];
        v8f_t pv = __builtin_amdgcn_wmma_f32_16x16x128_fp8_fp8(A, Bv, (short)0, zero, false, false);
        v8f_t pg = __builtin_amdgcn_wmma_f32_16x16x128_fp8_fp8(A, Bg, (short)0, zero, false, false);
#pragma unroll
        for (int r = 0; r < 8; ++r) {
            accV[r] += pv[r] * (sa[r] * sbv);
            accG[r] += pg[r] * (sa[r] * sbg);
        }
    }
    float* outBase = actbuf + ((size_t)e * T_TOK + t0) * I_DIM + i0 + (lane & 15);
    const int moff = (lane >> 4) << 3;
#pragma unroll
    for (int r = 0; r < 8; ++r) {
        float g = accG[r];
        float sg = g / (1.0f + __expf(-g));
        outBase[(size_t)(moff + r) * I_DIM] = sg * accV[r];
    }
}

__global__ void __launch_bounds__(128, 1)
moe_gemm2_kernel(const uint8_t* __restrict__ act2,
                 const float* __restrict__ a2s,
                 const uint8_t* __restrict__ w2T,
                 const float* __restrict__ w2s,
                 const float* __restrict__ Wr,
                 float* __restrict__ out) {
    const int lane = threadIdx.x & 31;
    const int wave = threadIdx.x >> 5;
    const int ht = blockIdx.x * 4 + wave;
    const int h0 = ht * 16;
    const int t0 = blockIdx.y * 16;
    const int hb = h0 >> 7;
    const int mbase = t0 + ((lane >> 4) << 3);

    v8f_t acc = {};
    const v8f_t zero = {};
#pragma unroll 1
    for (int e = 0; e < E_EXP; ++e) {
        float wrow[8];
        float wsum = 0.f;
#pragma unroll
        for (int r = 0; r < 8; ++r) {
            wrow[r] = Wr[(mbase + r) * E_EXP + e];
            wsum += fabsf(wrow[r]);
        }
        if (!__any(wsum != 0.0f)) continue;

        const uint8_t* aRow = act2 + ((size_t)e * T_TOK + t0 + (lane & 15)) * I_DIM;
        const uint8_t* bCol = w2T + (size_t)e * I_DIM * H_DIM + h0;
        const float* saBase = a2s + ((size_t)e * T_TOK + mbase) * KB_I;
        const float* sbBase = w2s + e * 48 + hb * 6;
        v8f_t acce = {};
#pragma unroll 1
        for (int kb = 0; kb < KB_I; ++kb) {
            if (kb + 1 < KB_I)
                __builtin_prefetch(bCol + (size_t)(kb + 1) * 128 * H_DIM + lane * H_DIM, 0, 1);
            v16i_t A = load_a_frag(aRow, kb * 128, lane);
            v16i_t B = load_b_frag(bCol, kb * 128, H_DIM, lane);
            float sb = sbBase[kb];
            float sa[8];
#pragma unroll
            for (int r = 0; r < 8; ++r) sa[r] = saBase[r * KB_I + kb];
            v8f_t p = __builtin_amdgcn_wmma_f32_16x16x128_fp8_fp8(A, B, (short)0, zero, false, false);
#pragma unroll
            for (int r = 0; r < 8; ++r) acce[r] += p[r] * (sa[r] * sb);
        }
#pragma unroll
        for (int r = 0; r < 8; ++r) acc[r] += wrow[r] * acce[r];
    }
    float* o = out + (size_t)t0 * H_DIM + h0 + (lane & 15);
    const int moff = (lane >> 4) << 3;
#pragma unroll
    for (int r = 0; r < 8; ++r)
        o[(size_t)(moff + r) * H_DIM] = acc[r];
}

extern "C" void kernel_launch(void* const* d_in, const int* in_sizes, int n_in,
                              void* d_out, int out_size, void* d_ws, size_t ws_size,
                              hipStream_t stream) {
    const float* hidden  = (const float*)d_in[0];
    const float* logits  = (const float*)d_in[1];
    const float* w13     = (const float*)d_in[2];
    const float* w13s    = (const float*)d_in[3];
    const float* w2      = (const float*)d_in[4];
    const float* w2s     = (const float*)d_in[5];
    float* out = (float*)d_out;

    uint8_t* ws = (uint8_t*)d_ws;
    size_t off = 0;
    float*   Wroute  = (float*)(ws + off);  off += (size_t)T_TOK * E_EXP * 4;
    uint8_t* act_fp8 = ws + off;            off += (size_t)T_TOK * H_DIM;
    float*   a_scale = (float*)(ws + off);  off += (size_t)T_TOK * KB_H * 4;
    uint8_t* w13T    = ws + off;            off += (size_t)E_EXP * H_DIM * I2_DIM;
    uint8_t* w2T     = ws + off;            off += (size_t)E_EXP * I_DIM * H_DIM;
    float*   actbuf  = (float*)(ws + off);  off += (size_t)E_EXP * T_TOK * I_DIM * 4;
    uint8_t* act2    = ws + off;            off += (size_t)E_EXP * T_TOK * I_DIM;
    float*   a2s     = (float*)(ws + off);  off += (size_t)E_EXP * T_TOK * KB_I * 4;
    (void)ws_size; (void)in_sizes; (void)n_in; (void)out_size;

    moe_router_kernel<<<(T_TOK + 255) / 256, 256, 0, stream>>>(logits, Wroute);

    {
        size_t n13 = (size_t)E_EXP * I2_DIM * H_DIM;
        cvt_w13_kernel<<<(unsigned)((n13 + 255) / 256), 256, 0, stream>>>(w13, w13T);
        size_t n2 = (size_t)E_EXP * H_DIM * I_DIM;
        cvt_w2_kernel<<<(unsigned)((n2 + 255) / 256), 256, 0, stream>>>(w2, w2T);
    }

    {
        int nrows = T_TOK * KB_H;
        quant_rows_kernel<<<(nrows + 7) / 8, 256, 0, stream>>>(hidden, act_fp8, a_scale, nrows);
    }

    moe_gemm1_kernel<<<dim3(I_DIM / 16 / 4, T_TOK / 16, E_EXP), 128, 0, stream>>>(
        act_fp8, a_scale, w13T, w13s, actbuf);

    {
        int nrows = E_EXP * T_TOK * KB_I;
        quant_rows_kernel<<<(nrows + 7) / 8, 256, 0, stream>>>(actbuf, act2, a2s, nrows);
    }

    moe_gemm2_kernel<<<dim3(H_DIM / 16 / 4, T_TOK / 16), 128, 0, stream>>>(
        act2, a2s, w2T, w2s, Wroute, out);
}
